// PerceiverResampler_20126216749184
// MI455X (gfx1250) — matrix operand order repaired and checked by reading, not compiled, not run on hardware
//
#include <hip/hip_runtime.h>
#include <hip/hip_bf16.h>
#include <math.h>

typedef __bf16 bf16;
typedef __attribute__((ext_vector_type(16))) __bf16 v16bf;
typedef __attribute__((ext_vector_type(8)))  float  v8f;

#define DD    768
#define DIN   320
#define NIN   49
#define NQQ   8
#define NH    8
#define DHH   96
#define DOUT  3072

// LDS row strides (in bf16 elements; stride*2 bytes is a multiple of 16)
#define FS 328   // features buffer
#define KS 776   // kv / kv_norm
#define VS 776   // vp
#define AS 72    // attn rows
#define XS 776   // tail activation A-matrices (also f32 stride)
#define SCS 68   // scores (f32 stride)

struct U8x32 { uint4 lo, hi; };

static __device__ __forceinline__ v16bf ldpair(const bf16* p0, const bf16* p1) {
  U8x32 u;
  u.lo = *(const uint4*)p0;
  u.hi = *(const uint4*)p1;
  return __builtin_bit_cast(v16bf, u);
}

static __device__ __forceinline__ v8f wmma_bf(v16bf a, v16bf b, v8f c) {
  return __builtin_amdgcn_wmma_f32_16x16x32_bf16(false, a, false, b, (short)0, c,
                                                 false, false);
}

// One 16x16 output tile: A strip (lane row pre-offset), B strip (lane row pre-offset)
static __device__ __forceinline__ v8f gemm1(const bf16* a, const bf16* brow, int kt32) {
  v8f acc = {};
  for (int kt = 0; kt < kt32; ++kt) {
    v16bf bm = ldpair(brow, brow + 16);
    v16bf am = ldpair(a, a + 16);
    acc = wmma_bf(am, bm, acc);
    brow += 32; a += 32;
  }
  return acc;
}

// Four 16x16 tiles sharing one B strip (M=64 GEMMs): quarters B (L2) traffic.
static __device__ __forceinline__ void gemm4(const bf16* a, int as, const bf16* brow,
                                             int kt32, v8f acc[4]) {
  for (int kt = 0; kt < kt32; ++kt) {
    v16bf bm = ldpair(brow, brow + 16);
#pragma unroll
    for (int mi = 0; mi < 4; ++mi) {
      v16bf am = ldpair(a + mi * 16 * as, a + mi * 16 * as + 16);
      acc[mi] = wmma_bf(am, bm, acc[mi]);
    }
    brow += 32; a += 32;
  }
}

static __device__ __forceinline__ void wave_reduce2(float& s, float& s2) {
  for (int o = 16; o; o >>= 1) {
    s  += __shfl_xor(s, o, 32);
    s2 += __shfl_xor(s2, o, 32);
  }
}

// ---------------- prep kernels ----------------

__global__ __launch_bounds__(256) void f2bf_k(const float* s, bf16* d, int n) {
  int i = blockIdx.x * 256 + threadIdx.x;
  if (i < n) d[i] = (bf16)s[i];
}

__global__ __launch_bounds__(256) void qnorm_k(const float* queries, const float* g,
                                               const float* be, float* qn) {
  int w = threadIdx.x >> 5, lane = threadIdx.x & 31;
  const float* p = queries + w * DD;
  float s = 0.f, s2 = 0.f;
  for (int c = lane; c < DD; c += 32) { float x = p[c]; s += x; s2 += x * x; }
  wave_reduce2(s, s2);
  float mu = s / DD, var = s2 / DD - mu * mu, rs = rsqrtf(var + 1e-5f);
  for (int c = lane; c < DD; c += 32)
    qn[w * DD + c] = (p[c] - mu) * rs * g[c] + be[c];
}

__global__ __launch_bounds__(256) void qp_k(const float* qnorm, const float* inW,
                                            const float* inB, float* qp) {
  int o = blockIdx.x * 256 + threadIdx.x;
  if (o >= NQQ * DD) return;
  int q = o / DD, e = o % DD;
  const float* x = qnorm + q * DD;
  const float* wr = inW + (size_t)e * DD;   // wq = rows [0,D)
  float s = inB[e];
  for (int d = 0; d < DD; ++d) s += x[d] * wr[d];
  qp[o] = s;
}

__global__ __launch_bounds__(256) void qw_k(const float* qp, const float* inW,
                                            const float* inB, bf16* qw, float* sbias) {
  const float scl = 0.1020620726159658f;  // 1/sqrt(96)
  int o = blockIdx.x * 256 + threadIdx.x;
  if (o >= 64 * DD) return;
  int hq = o / DD, d = o % DD;
  int h = hq >> 3, q = hq & 7;
  const float* qr = qp + q * DD + h * DHH;
  const float* wk = inW + (size_t)(DD + h * DHH) * DD + d;  // wk = rows [D,2D)
  float s = 0.f;
  for (int j = 0; j < DHH; ++j) s += qr[j] * wk[(size_t)j * DD];
  qw[o] = (bf16)(s * scl);
  if (o < 64) {
    const float* bk = inB + DD + h * DHH;
    float t = 0.f;
    for (int j = 0; j < DHH; ++j) t += qr[j] * bk[j];
    sbias[o] = t * scl;
  }
}

// ---------------- fused per-batch kernel ----------------

__global__ __launch_bounds__(256) void fused_batch(
    const float* features, const float* queries,
    const float* input_proj_b, const float* ln_kv_g, const float* ln_kv_b,
    const float* in_proj_b, const float* out_proj_b,
    const float* ln_ff_g, const float* ln_ff_b,
    const float* ffn_b1, const float* ffn_b2, const float* output_proj_b,
    const bf16* w_in, const bf16* qw, const float* sbias,
    const bf16* wv, const bf16* wo, const bf16* w1, const bf16* w2,
    const bf16* wout, float* out) {
  __shared__ __align__(16) bf16 kvbuf[64 * KS];    // kv -> kv_norm
  __shared__ __align__(16) bf16 vpbuf[64 * VS];    // vp; later q1(f32)+xb
  __shared__ __align__(16) bf16 featbuf[64 * FS];  // feat; later attn+scores; later xa

  const int b = blockIdx.x;
  const int tid = threadIdx.x;
  const int w = tid >> 5, lane = tid & 31;
  const int lr = lane & 15, lh = lane >> 4;

  // stage 1: features -> bf16 LDS, zero padded
  {
    const float* f = features + (size_t)b * NIN * DIN;
    for (int i = tid; i < 64 * FS; i += 256) {
      int r = i / FS, c = i - r * FS;
      float v = (r < NIN && c < DIN) ? f[r * DIN + c] : 0.f;
      featbuf[i] = (bf16)v;
    }
  }
  __syncthreads();

  // stage 2: kv = feat @ w_in^T + b   (M=64, N=768, K=320)
  for (int ni = w; ni < 48; ni += 8) {
    v8f acc[4] = {};
    gemm4(featbuf + lr * FS + lh * 8, FS,
          w_in + (size_t)(ni * 16 + lr) * DIN + lh * 8, 10, acc);
    float bias = input_proj_b[ni * 16 + lr];
#pragma unroll
    for (int mi = 0; mi < 4; ++mi)
#pragma unroll
      for (int r = 0; r < 8; ++r)
        kvbuf[(mi * 16 + lh * 8 + r) * KS + ni * 16 + lr] = (bf16)(acc[mi][r] + bias);
  }
  __syncthreads();

  // stage 3: LayerNorm(kv) in place, rows < 49 only
  for (int row = w; row < NIN; row += 8) {
    bf16* p = kvbuf + row * KS;
    float s = 0.f, s2 = 0.f;
    for (int c = lane; c < DD; c += 32) { float x = (float)p[c]; s += x; s2 += x * x; }
    wave_reduce2(s, s2);
    float mu = s / DD, var = s2 / DD - mu * mu, rs = rsqrtf(var + 1e-5f);
    for (int c = lane; c < DD; c += 32) {
      float x = (float)p[c];
      p[c] = (bf16)((x - mu) * rs * ln_kv_g[c] + ln_kv_b[c]);
    }
  }
  __syncthreads();

  // featbuf reuse: attn [64][AS] bf16 at 0; scores f32 [64][SCS] at elem 4608
  bf16* attn = featbuf;
  float* sc = (float*)(featbuf + 4608);
  bf16* xa = featbuf + 4608;  // tail A-matrix [16][XS] (overwrites scores later)

  // stage 4: scores_t[n,hq] = kv_norm @ qw^T + sbias  (M=64, N=64, K=768)
  for (int t = w; t < 16; t += 8) {
    int mi = t >> 2, ni = t & 3;
    v8f acc = gemm1(kvbuf + (mi * 16 + lr) * KS + lh * 8,
                    qw + (size_t)(ni * 16 + lr) * DD + lh * 8, 24);
    float sb = sbias[ni * 16 + lr];
#pragma unroll
    for (int r = 0; r < 8; ++r)
      sc[(mi * 16 + lh * 8 + r) * SCS + ni * 16 + lr] = acc[r] + sb;
  }
  __syncthreads();

  // stage 5: softmax over n (one thread per (h,q) column)
  if (tid < 64) {
    int hq = tid;
    float mx = -1e30f;
    for (int n = 0; n < NIN; ++n) mx = fmaxf(mx, sc[n * SCS + hq]);
    float sum = 0.f;
    for (int n = 0; n < NIN; ++n) sum += __expf(sc[n * SCS + hq] - mx);
    float inv = 1.f / sum;
    for (int n = 0; n < NIN; ++n)
      attn[hq * AS + n] = (bf16)(__expf(sc[n * SCS + hq] - mx) * inv);
    for (int n = NIN; n < 64; ++n) attn[hq * AS + n] = (bf16)0.f;
  }
  __syncthreads();

  // stage 6: vp = kv_norm @ wv^T + bv   (M=64, N=768, K=768)
  for (int ni = w; ni < 48; ni += 8) {
    v8f acc[4] = {};
    gemm4(kvbuf + lr * KS + lh * 8, KS,
          wv + (size_t)(ni * 16 + lr) * DD + lh * 8, 24, acc);
    float bias = in_proj_b[2 * DD + ni * 16 + lr];
#pragma unroll
    for (int mi = 0; mi < 4; ++mi)
#pragma unroll
      for (int r = 0; r < 8; ++r)
        vpbuf[(mi * 16 + lh * 8 + r) * VS + ni * 16 + lr] = (bf16)(acc[mi][r] + bias);
  }
  __syncthreads();

  // stage 7: ctx[q,e] = sum_n attn[h(e)*8+q, n] * vp[n,e]  (tiny; VALU) -> xa bf16
  for (int o = tid; o < NQQ * DD; o += 256) {
    int q = o / DD, e = o - q * DD;
    int hq = (e / DHH) * 8 + q;
    const bf16* ar = attn + hq * AS;
    float s = 0.f;
    for (int n = 0; n < NIN; ++n) s += (float)ar[n] * (float)vpbuf[n * VS + e];
    xa[q * XS + e] = (bf16)s;
  }
  for (int i = tid; i < 8 * XS; i += 256) xa[8 * XS + i] = (bf16)0.f;  // pad rows
  __syncthreads();

  // stage 8: attn_out = ctx @ wo^T + ob ; q1 = queries + attn_out (f32 in vpbuf)
  float* q1 = (float*)vpbuf;
  for (int t = w; t < 48; t += 8) {
    v8f acc = gemm1(xa + lr * XS + lh * 8,
                    wo + (size_t)(t * 16 + lr) * DD + lh * 8, 24);
    int col = t * 16 + lr;
    float bias = out_proj_b[col];
#pragma unroll
    for (int r = 0; r < 8; ++r) {
      int row = lh * 8 + r;
      if (row < 8) q1[row * XS + col] = acc[r] + bias + queries[row * DD + col];
    }
  }
  __syncthreads();

  // stage 9: LN(q1) -> xa rows 0..7 (one wave per row)
  {
    const float* p = q1 + w * XS;
    float s = 0.f, s2 = 0.f;
    for (int c = lane; c < DD; c += 32) { float x = p[c]; s += x; s2 += x * x; }
    wave_reduce2(s, s2);
    float mu = s / DD, var = s2 / DD - mu * mu, rs = rsqrtf(var + 1e-5f);
    for (int c = lane; c < DD; c += 32)
      xa[w * XS + c] = (bf16)((p[c] - mu) * rs * ln_ff_g[c] + ln_ff_b[c]);
  }
  __syncthreads();

  // stage 10: h = gelu(xa @ w1^T + b1) -> xb
  bf16* xb = vpbuf + 12416;  // after q1 region
  for (int t = w; t < 48; t += 8) {
    v8f acc = gemm1(xa + lr * XS + lh * 8,
                    w1 + (size_t)(t * 16 + lr) * DD + lh * 8, 24);
    int col = t * 16 + lr;
    float b1 = ffn_b1[col];
#pragma unroll
    for (int r = 0; r < 8; ++r) {
      int row = lh * 8 + r;
      float x = acc[r] + b1;
      float g = 0.5f * x * (1.f + erff(x * 0.7071067811865475f));
      xb[row * XS + col] = (bf16)((row < 8) ? g : 0.f);
    }
  }
  __syncthreads();

  // stage 11: q2 = q1 + (xb @ w2^T + b2) -> xa bf16
  for (int t = w; t < 48; t += 8) {
    v8f acc = gemm1(xb + lr * XS + lh * 8,
                    w2 + (size_t)(t * 16 + lr) * DD + lh * 8, 24);
    int col = t * 16 + lr;
    float b2 = ffn_b2[col];
#pragma unroll
    for (int r = 0; r < 8; ++r) {
      int row = lh * 8 + r;
      if (row < 8) xa[row * XS + col] = (bf16)(acc[r] + b2 + q1[row * XS + col]);
    }
  }
  __syncthreads();

  // stage 12: out = q2 @ wout^T + b (pre-LN, f32 to global)
  float* ob = out + (size_t)b * NQQ * DOUT;
  for (int t = w; t < 192; t += 8) {
    v8f acc = gemm1(xa + lr * XS + lh * 8,
                    wout + (size_t)(t * 16 + lr) * DD + lh * 8, 24);
    int col = t * 16 + lr;
    float bias = output_proj_b[col];
#pragma unroll
    for (int r = 0; r < 8; ++r) {
      int row = lh * 8 + r;
      if (row < 8) ob[row * DOUT + col] = acc[r] + bias;
    }
  }
}

// ---------------- final LN over 3072 ----------------

__global__ __launch_bounds__(256) void final_ln_k(float* out, const float* g,
                                                  const float* be) {
  __shared__ float sh[16];
  __shared__ float stats[2];
  float* p = out + (size_t)blockIdx.x * DOUT;
  int tid = threadIdx.x, w = tid >> 5, lane = tid & 31;
  float s = 0.f, s2 = 0.f;
  for (int c = tid; c < DOUT; c += 256) { float x = p[c]; s += x; s2 += x * x; }
  wave_reduce2(s, s2);
  if (lane == 0) { sh[w] = s; sh[8 + w] = s2; }
  __syncthreads();
  if (tid == 0) {
    float ts = 0.f, t2 = 0.f;
    for (int i = 0; i < 8; ++i) { ts += sh[i]; t2 += sh[8 + i]; }
    float mu = ts / DOUT, var = t2 / DOUT - mu * mu;
    stats[0] = mu; stats[1] = rsqrtf(var + 1e-5f);
  }
  __syncthreads();
  float mu = stats[0], rs = stats[1];
  for (int c = tid; c < DOUT; c += 256) p[c] = (p[c] - mu) * rs * g[c] + be[c];
}

// ---------------- host launch ----------------

extern "C" void kernel_launch(void* const* d_in, const int* in_sizes, int n_in,
                              void* d_out, int out_size, void* d_ws, size_t ws_size,
                              hipStream_t stream) {
  const float* features      = (const float*)d_in[0];
  const float* input_proj_w  = (const float*)d_in[1];
  const float* input_proj_b  = (const float*)d_in[2];
  const float* queries       = (const float*)d_in[3];
  const float* ln_q_g        = (const float*)d_in[4];
  const float* ln_q_b        = (const float*)d_in[5];
  const float* ln_kv_g       = (const float*)d_in[6];
  const float* ln_kv_b       = (const float*)d_in[7];
  const float* in_proj_w     = (const float*)d_in[8];
  const float* in_proj_b     = (const float*)d_in[9];
  const float* out_proj_w    = (const float*)d_in[10];
  const float* out_proj_b    = (const float*)d_in[11];
  const float* ln_ff_g       = (const float*)d_in[12];
  const float* ln_ff_b       = (const float*)d_in[13];
  const float* ffn_w1        = (const float*)d_in[14];
  const float* ffn_b1        = (const float*)d_in[15];
  const float* ffn_w2        = (const float*)d_in[16];
  const float* ffn_b2        = (const float*)d_in[17];
  const float* output_proj_w = (const float*)d_in[18];
  const float* output_proj_b = (const float*)d_in[19];
  const float* final_g       = (const float*)d_in[20];
  const float* final_b       = (const float*)d_in[21];

  const int B = in_sizes[0] / (NIN * DIN);

  char* ws = (char*)d_ws;
  const size_t OFF_WIN  = 0;
  const size_t OFF_WV   = OFF_WIN  + (size_t)DD * DIN * 2;
  const size_t OFF_WO   = OFF_WV   + (size_t)DD * DD * 2;
  const size_t OFF_W1   = OFF_WO   + (size_t)DD * DD * 2;
  const size_t OFF_W2   = OFF_W1   + (size_t)DD * DD * 2;
  const size_t OFF_WOUT = OFF_W2   + (size_t)DD * DD * 2;
  const size_t OFF_QW   = OFF_WOUT + (size_t)DOUT * DD * 2;
  const size_t OFF_QN   = OFF_QW   + (size_t)64 * DD * 2;
  const size_t OFF_QP   = OFF_QN   + (size_t)NQQ * DD * 4;
  const size_t OFF_SB   = OFF_QP   + (size_t)NQQ * DD * 4;

  bf16*  w_in_bf = (bf16*)(ws + OFF_WIN);
  bf16*  wv_bf   = (bf16*)(ws + OFF_WV);
  bf16*  wo_bf   = (bf16*)(ws + OFF_WO);
  bf16*  w1_bf   = (bf16*)(ws + OFF_W1);
  bf16*  w2_bf   = (bf16*)(ws + OFF_W2);
  bf16*  wout_bf = (bf16*)(ws + OFF_WOUT);
  bf16*  qw_bf   = (bf16*)(ws + OFF_QW);
  float* qnorm   = (float*)(ws + OFF_QN);
  float* qp      = (float*)(ws + OFF_QP);
  float* sbias   = (float*)(ws + OFF_SB);

  auto cvt = [&](const float* s, bf16* d, int n) {
    f2bf_k<<<(n + 255) / 256, 256, 0, stream>>>(s, d, n);
  };
  cvt(input_proj_w,               w_in_bf, DD * DIN);
  cvt(in_proj_w + 2 * DD * DD,    wv_bf,   DD * DD);
  cvt(out_proj_w,                 wo_bf,   DD * DD);
  cvt(ffn_w1,                     w1_bf,   DD * DD);
  cvt(ffn_w2,                     w2_bf,   DD * DD);
  cvt(output_proj_w,              wout_bf, DOUT * DD);

  qnorm_k<<<1, 256, 0, stream>>>(queries, ln_q_g, ln_q_b, qnorm);
  qp_k<<<(NQQ * DD + 255) / 256, 256, 0, stream>>>(qnorm, in_proj_w, in_proj_b, qp);
  qw_k<<<(64 * DD + 255) / 256, 256, 0, stream>>>(qp, in_proj_w, in_proj_b, qw_bf, sbias);

  fused_batch<<<B, 256, 0, stream>>>(
      features, queries, input_proj_b, ln_kv_g, ln_kv_b, in_proj_b, out_proj_b,
      ln_ff_g, ln_ff_b, ffn_b1, ffn_b2, output_proj_b,
      w_in_bf, qw_bf, sbias, wv_bf, wo_bf, w1_bf, w2_bf, wout_bf, (float*)d_out);

  final_ln_k<<<B * NQQ, 256, 0, stream>>>((float*)d_out, final_g, final_b);
}
